// AttentionLayer_72310069395789
// MI455X (gfx1250) — matrix operand order repaired and checked by reading, not compiled, not run on hardware
//
#include <hip/hip_runtime.h>
#include <hip/hip_bf16.h>
#include <math.h>

typedef __attribute__((ext_vector_type(16))) __bf16 v16bf;
typedef __attribute__((ext_vector_type(8)))  float  v8f;

#define BB 64
#define TT 4096
#define DD 256
#define UU 64

// ---------------------------------------------------------------------------
// Pass 1: logits[b,t] = tanh(x[b,t,:] @ W + bias) . u      (bf16 WMMA GEMM)
// 256 threads = 8 waves; each wave does 32 rows (2 M-tiles) x 64 U via
// 64 wmma ops, with each LDS B-fragment feeding 2 WMMAs.
// Raw logits are written into the attention-weights region of d_out.
// ---------------------------------------------------------------------------
__global__ __launch_bounds__(256) void logits_wmma_kernel(
    const float* __restrict__ x, const float* __restrict__ W,
    const float* __restrict__ bias, const float* __restrict__ uvec,
    float* __restrict__ out)
{
    // W is [D,U] row-major in global; store transposed Wt[u][k] as bf16 in LDS
    __shared__ __align__(32) __bf16 sW[UU * DD];   // 32 KB
    const int tid = threadIdx.x;
    for (int i = tid; i < DD * UU; i += 256) {
        int k  = i >> 6;      // 0..255  (coalesced global read: i = k*64 + u)
        int uu = i & 63;
        sW[uu * DD + k] = (__bf16)W[i];
    }

    const int lane = tid & 31;
    const int wave = tid >> 5;
    const int col  = lane & 15;      // A-row index within tile / C column
    const int hi   = lane >> 4;      // lane half
    const int rowBase = blockIdx.x * 256 + wave * 32;
    const float* xrow0 = x + (size_t)(rowBase + col) * DD;        // M-tile 0
    const float* xrow1 = x + (size_t)(rowBase + 16 + col) * DD;   // M-tile 1
    const int koff = hi * 16;        // A order used here (contiguous): halves hold K 0..15 / 16..31

    // hoist bias / u loads (overlap with LDS fill)
    float bv[4], uv[4];
    #pragma unroll
    for (int n = 0; n < 4; ++n) {
        bv[n] = bias[n * 16 + col];
        uv[n] = uvec[n * 16 + col];
    }
    __syncthreads();

    v8f acc[8] = {};                 // [n][m] : 4 N-tiles x 2 M-tiles

    #pragma unroll
    for (int kt = 0; kt < 8; ++kt) {             // K = 256 in steps of 32
        const float* p0 = xrow0 + kt * 32 + koff;
        const float* p1 = xrow1 + kt * 32 + koff;
        float4 f0 = *(const float4*)(p0);        // K = koff+0..3
        float4 f1 = *(const float4*)(p0 + 4);    // K = koff+4..7
        float4 f2 = *(const float4*)(p0 + 8);    // K = koff+8..11
        float4 f3 = *(const float4*)(p0 + 12);   // K = koff+12..15
        float4 g0 = *(const float4*)(p1);
        float4 g1 = *(const float4*)(p1 + 4);
        float4 g2 = *(const float4*)(p1 + 8);
        float4 g3 = *(const float4*)(p1 + 12);
        v16bf a0, a1;
        a0[0]=(__bf16)f0.x; a0[1]=(__bf16)f0.y; a0[2]=(__bf16)f0.z; a0[3]=(__bf16)f0.w;
        a0[4]=(__bf16)f1.x; a0[5]=(__bf16)f1.y; a0[6]=(__bf16)f1.z; a0[7]=(__bf16)f1.w;
        a0[8]=(__bf16)f2.x; a0[9]=(__bf16)f2.y; a0[10]=(__bf16)f2.z; a0[11]=(__bf16)f2.w;
        a0[12]=(__bf16)f3.x; a0[13]=(__bf16)f3.y; a0[14]=(__bf16)f3.z; a0[15]=(__bf16)f3.w;
        a1[0]=(__bf16)g0.x; a1[1]=(__bf16)g0.y; a1[2]=(__bf16)g0.z; a1[3]=(__bf16)g0.w;
        a1[4]=(__bf16)g1.x; a1[5]=(__bf16)g1.y; a1[6]=(__bf16)g1.z; a1[7]=(__bf16)g1.w;
        a1[8]=(__bf16)g2.x; a1[9]=(__bf16)g2.y; a1[10]=(__bf16)g2.z; a1[11]=(__bf16)g2.w;
        a1[12]=(__bf16)g3.x; a1[13]=(__bf16)g3.y; a1[14]=(__bf16)g3.z; a1[15]=(__bf16)g3.w;

        const int kb = kt * 32 + hi * 16;        // B order matches A (contiguous): halves hold K 0..15 / 16..31
        #pragma unroll
        for (int n = 0; n < 4; ++n) {
            const v16bf bfr = *(const v16bf*)(&sW[(n * 16 + col) * DD + kb]);
            acc[n * 2 + 0] = __builtin_amdgcn_wmma_f32_16x16x32_bf16(
                false, a0, false, bfr, (short)0, acc[n * 2 + 0], false, false);
            acc[n * 2 + 1] = __builtin_amdgcn_wmma_f32_16x16x32_bf16(
                false, a1, false, bfr, (short)0, acc[n * 2 + 1], false, false);
        }
    }

    // Epilogue: +bias, tanh, *u, then reduce over the 64 U columns.
    // C layout: lane column = col; VGPR r holds row r (lo half) / row 8+r (hi half).
    float part0[8], part1[8];
    #pragma unroll
    for (int r = 0; r < 8; ++r) { part0[r] = 0.0f; part1[r] = 0.0f; }
    #pragma unroll
    for (int n = 0; n < 4; ++n) {
        #pragma unroll
        for (int r = 0; r < 8; ++r) {
            part0[r] += tanhf(acc[n * 2 + 0][r] + bv[n]) * uv[n];
            part1[r] += tanhf(acc[n * 2 + 1][r] + bv[n]) * uv[n];
        }
    }
    // reduce across the 16 lanes of each half (rows differ per half -> masks <= 8)
    #pragma unroll
    for (int m = 1; m < 16; m <<= 1) {
        #pragma unroll
        for (int r = 0; r < 8; ++r) {
            part0[r] += __shfl_xor(part0[r], m, 32);
            part1[r] += __shfl_xor(part1[r], m, 32);
        }
    }
    if (col == 0) {
        float* lg = out + (size_t)BB * DD;       // logits live in attn-weight slot
        const int rb = rowBase + hi * 8;
        #pragma unroll
        for (int r = 0; r < 8; ++r) {
            lg[rb + r]      = part0[r];
            lg[rb + 16 + r] = part1[r];
        }
    }
}

// ---------------------------------------------------------------------------
// Pass 2: per-batch softmax over T (in place) + zero the context row.
// ---------------------------------------------------------------------------
__global__ __launch_bounds__(256) void softmax_kernel(float* __restrict__ out)
{
    const int b   = blockIdx.x;
    const int tid = threadIdx.x;
    float* lg = out + (size_t)BB * DD + (size_t)b * TT;

    out[b * DD + tid] = 0.0f;        // zero context[b,:] for pass-3 atomics

    __shared__ float sred[8];
    float lmax = -3.402823466e38f;
    for (int t = tid; t < TT; t += 256) lmax = fmaxf(lmax, lg[t]);
    #pragma unroll
    for (int m = 16; m >= 1; m >>= 1) lmax = fmaxf(lmax, __shfl_xor(lmax, m, 32));
    if ((tid & 31) == 0) sred[tid >> 5] = lmax;
    __syncthreads();
    float gmax = sred[0];
    #pragma unroll
    for (int i = 1; i < 8; ++i) gmax = fmaxf(gmax, sred[i]);
    __syncthreads();

    float lsum = 0.0f;
    for (int t = tid; t < TT; t += 256) lsum += expf(lg[t] - gmax);
    #pragma unroll
    for (int m = 16; m >= 1; m >>= 1) lsum += __shfl_xor(lsum, m, 32);
    if ((tid & 31) == 0) sred[tid >> 5] = lsum;
    __syncthreads();
    float gsum = 0.0f;
    #pragma unroll
    for (int i = 0; i < 8; ++i) gsum += sred[i];
    const float inv = 1.0f / gsum;

    for (int t = tid; t < TT; t += 256) lg[t] = expf(lg[t] - gmax) * inv;
}

// ---------------------------------------------------------------------------
// Pass 3: context[b,d] = sum_t x[b,t,d] * aw[b,t].  Grid (B, 16 T-chunks),
// 256 threads = one per d. Coalesced 1KB rows; weights staged in LDS.
// ---------------------------------------------------------------------------
__global__ __launch_bounds__(256) void context_kernel(
    const float* __restrict__ x, float* __restrict__ out)
{
    const int b   = blockIdx.x;
    const int c   = blockIdx.y;          // 16 chunks of 256 timesteps
    const int tid = threadIdx.x;         // d index

    __shared__ float saw[256];
    const float* aw = out + (size_t)BB * DD + (size_t)b * TT + (size_t)c * 256;
    saw[tid] = aw[tid];
    __syncthreads();

    const float* xp = x + ((size_t)b * TT + (size_t)c * 256) * DD + tid;
    float s = 0.0f;
    #pragma unroll 8
    for (int t = 0; t < 256; ++t)
        s = fmaf(xp[(size_t)t * DD], saw[t], s);

    atomicAdd(out + b * DD + tid, s);
}

// ---------------------------------------------------------------------------
extern "C" void kernel_launch(void* const* d_in, const int* in_sizes, int n_in,
                              void* d_out, int out_size, void* d_ws, size_t ws_size,
                              hipStream_t stream) {
    (void)in_sizes; (void)n_in; (void)out_size; (void)d_ws; (void)ws_size;
    const float* x    = (const float*)d_in[0];
    const float* W    = (const float*)d_in[1];
    const float* bias = (const float*)d_in[2];
    const float* uvec = (const float*)d_in[3];
    float* out = (float*)d_out;

    logits_wmma_kernel<<<dim3((BB * TT) / 256), 256, 0, stream>>>(x, W, bias, uvec, out);
    softmax_kernel<<<dim3(BB), 256, 0, stream>>>(out);
    context_kernel<<<dim3(BB, 16), 256, 0, stream>>>(x, out);
}
